// SpikingTemporalRNN_51908974739528
// MI455X (gfx1250) — matrix operand order repaired and checked by reading, not compiled, not run on hardware
//
#include <hip/hip_runtime.h>

typedef __bf16 bf16_t;
typedef bf16_t bf16x8  __attribute__((ext_vector_type(8)));
typedef bf16_t bf16x16 __attribute__((ext_vector_type(16)));
typedef float  v8f     __attribute__((ext_vector_type(8)));
typedef float  v2f     __attribute__((ext_vector_type(2)));

#define BATCH   32
#define TSTEPS  2048
#define INDIM   512
#define HDIM    256
#define BT      (BATCH * TSTEPS)
#define BETA    0.85f
#define THR     1.0f

// spike LDS row stride in bytes: 256 bf16 = 512B + 16B pad -> conflict-free ds_load_b128
#define SPK_RS    528
#define SPK_BYTES (16 * SPK_RS)

// ---------------------------------------------------------------------------
// Kernel 0: convert V (HDIM x HDIM f32, row-major) to bf16 (RNE).
// Spikes are exactly representable in bf16, so only V is rounded.
// ---------------------------------------------------------------------------
__global__ void snn_prep_vbf(const float* __restrict__ V, bf16_t* __restrict__ Vbf) {
    int i = blockIdx.x * 256 + threadIdx.x;
    Vbf[i] = (bf16_t)V[i];
}

// ---------------------------------------------------------------------------
// Kernel 1: proj = x @ W_in^T + b_in  using f32 WMMA 16x16x4 (exact f32 math).
// One wave per 16-row strip of M=B*T, computing the full N=256 in registers.
// A-frag (16x4 f32): lane<16 -> K={0,1}, lane>=16 -> K={2,3}, M = lane&15.
// B-frag (4x16 f32): B[K][N] = W_in[n0+N][k0+K]; lane supplies N=lane&15,
//                    K = 2*(lane>>4) + {0,1}.
// ---------------------------------------------------------------------------
__global__ void __launch_bounds__(32) snn_proj_wmma(
        const float* __restrict__ x,
        const float* __restrict__ Win,
        const float* __restrict__ bin,
        float* __restrict__ proj) {
    const int lane = threadIdx.x & 31;
    const int l16  = lane & 15;
    const int hi   = lane >> 4;
    const long r0  = (long)blockIdx.x * 16;

    v8f acc[16];
#pragma unroll
    for (int nt = 0; nt < 16; ++nt)
#pragma unroll
        for (int j = 0; j < 8; ++j) acc[nt][j] = 0.0f;

    const float* xrow = x + (r0 + l16) * INDIM + 2 * hi;

#pragma unroll 1
    for (int k = 0; k < INDIM; k += 4) {
        v2f a = *(const v2f*)(xrow + k);
#pragma unroll
        for (int nt = 0; nt < 16; ++nt) {
            v2f b = *(const v2f*)(Win + (nt * 16 + l16) * INDIM + k + 2 * hi);
            acc[nt] = __builtin_amdgcn_wmma_f32_16x16x4_f32(
                false, a, false, b, (short)0, acc[nt], false, false);
        }
    }

    // C/D layout: VGPR i -> row M = i + 8*(lane>>4), col N = lane&15.
#pragma unroll
    for (int nt = 0; nt < 16; ++nt) {
        const int col = nt * 16 + l16;
        const float bias = bin[col];
#pragma unroll
        for (int i = 0; i < 8; ++i) {
            const long row = r0 + i + 8 * hi;
            proj[row * HDIM + col] = acc[nt][i] + bias;
        }
    }
}

// ---------------------------------------------------------------------------
// Kernel 2: the sequential LIF scan. 2 workgroups x 16 batch rows.
// rec = spk @ V^T via bf16 WMMA 16x16x32 (f32 accum); V fragments persist in
// VGPRs for all 2048 steps; spikes round-trip through padded LDS; proj is
// prefetched one step ahead.
// ---------------------------------------------------------------------------
__global__ void __launch_bounds__(256, 1) snn_scan(
        const float* __restrict__ proj,
        const bf16_t* __restrict__ Vbf,
        const float* __restrict__ brec,
        float* __restrict__ out) {
    extern __shared__ char smem[];   // SPK_BYTES: spk[16][SPK_RS bytes]

    const int tid  = threadIdx.x;
    const int lane = tid & 31;
    const int l16  = lane & 15;
    const int hi   = lane >> 4;
    const int wave = tid >> 5;
    const long b0  = (long)blockIdx.x * 16;

    // zero spike buffer (spk = 0 at t = 0)
    for (int i = tid; i < SPK_BYTES / 4; i += 256)
        ((unsigned*)smem)[i] = 0u;

    // --- persistent B fragments: B[K][N] = V[n][k], bf16, 2 n-tiles x 8 k-steps
    bf16x16 Bv[2][8];
    float   brv[2];
#pragma unroll
    for (int t2 = 0; t2 < 2; ++t2) {
        const int nt = 2 * wave + t2;
        const int n  = nt * 16 + l16;
        brv[t2] = brec[n];
#pragma unroll
        for (int kk = 0; kk < 8; ++kk) {
            // lane<16 supplies K = kk*32 + 0..15 ; lane>=16 supplies K = kk*32 + 16..31
            const bf16_t* p = Vbf + n * HDIM + kk * 32 + 16 * hi;
            bf16x8 blo = *(const bf16x8*)p;
            bf16x8 bhi = *(const bf16x8*)(p + 8);
            bf16x16 f;
#pragma unroll
            for (int j = 0; j < 8; ++j) { f[j] = blo[j]; f[j + 8] = bhi[j]; }
            Bv[t2][kk] = f;
        }
    }

    // persistent state (C/D layout: slot (t2,i) -> m = i + 8*hi, n = nt*16 + l16)
    v8f memv[2], spkv[2];
#pragma unroll
    for (int t2 = 0; t2 < 2; ++t2)
#pragma unroll
        for (int j = 0; j < 8; ++j) { memv[t2][j] = 0.0f; spkv[t2][j] = 0.0f; }

    // element offsets into proj/out at t=0: ((b0+m)*T + t)*H + n
    long pbase[2][8];
#pragma unroll
    for (int t2 = 0; t2 < 2; ++t2) {
        const int n = (2 * wave + t2) * 16 + l16;
#pragma unroll
        for (int i = 0; i < 8; ++i)
            pbase[t2][i] = ((b0 + i + 8 * hi) * (long)TSTEPS) * HDIM + n;
    }

    // prefetch proj for t = 0
    float prc[2][8], prn[2][8];
#pragma unroll
    for (int t2 = 0; t2 < 2; ++t2)
#pragma unroll
        for (int i = 0; i < 8; ++i) prc[t2][i] = proj[pbase[t2][i]];

    __syncthreads();   // spike buffer zeroed

#pragma unroll 1
    for (int t = 0; t < TSTEPS; ++t) {
        // prefetch proj for t+1 (overlaps with WMMA block)
        const int tn = (t + 1 < TSTEPS) ? (t + 1) : t;
#pragma unroll
        for (int t2 = 0; t2 < 2; ++t2)
#pragma unroll
            for (int i = 0; i < 8; ++i)
                prn[t2][i] = proj[pbase[t2][i] + (long)tn * HDIM];

        // A fragments from spike LDS (contiguous k order, same as B):
        // lane<16: K = kk*32 + 0..15; lane>=16: K = kk*32 + 16..31; M = lane&15
        bf16x16 Af[8];
#pragma unroll
        for (int kk = 0; kk < 8; ++kk) {
            const char* pa = smem + l16 * SPK_RS + (kk * 32 + 16 * hi) * 2;
            bf16x8 alo = *(const bf16x8*)pa;
            bf16x8 ahi = *(const bf16x8*)(pa + 16);
            bf16x16 f;
#pragma unroll
            for (int j = 0; j < 8; ++j) { f[j] = alo[j]; f[j + 8] = ahi[j]; }
            Af[kk] = f;
        }

        // rec = spk @ V^T + b_rec
        v8f c[2];
#pragma unroll
        for (int t2 = 0; t2 < 2; ++t2)
#pragma unroll
            for (int j = 0; j < 8; ++j) c[t2][j] = brv[t2];

#pragma unroll
        for (int kk = 0; kk < 8; ++kk)
#pragma unroll
            for (int t2 = 0; t2 < 2; ++t2)
                c[t2] = __builtin_amdgcn_wmma_f32_16x16x32_bf16(
                    false, Af[kk], false, Bv[t2][kk], (short)0, c[t2], false, false);

        __syncthreads();   // all spike reads for step t complete

        // LIF update + emit spikes
#pragma unroll
        for (int t2 = 0; t2 < 2; ++t2) {
            const int n = (2 * wave + t2) * 16 + l16;
#pragma unroll
            for (int i = 0; i < 8; ++i) {
                const float m_old = memv[t2][i];
                const float reset = (m_old > THR) ? 1.0f : 0.0f;
                const float m_new = BETA * m_old + prc[t2][i] + spkv[t2][i]
                                    + c[t2][i] - reset * THR;
                const float s_new = ((m_new - THR) > 0.0f) ? 1.0f : 0.0f;
                memv[t2][i] = m_new;
                spkv[t2][i] = s_new;
                out[pbase[t2][i] + (long)t * HDIM] = s_new;
                const int m = i + 8 * hi;
                *(bf16_t*)(smem + m * SPK_RS + n * 2) = (bf16_t)s_new;  // exact 0/1
            }
        }

        // roll prefetch buffer
#pragma unroll
        for (int t2 = 0; t2 < 2; ++t2)
#pragma unroll
            for (int i = 0; i < 8; ++i) prc[t2][i] = prn[t2][i];

        __syncthreads();   // new spikes visible for step t+1
    }
}

// ---------------------------------------------------------------------------
// launch: prep -> proj GEMM -> sequential scan (same stream => ordered)
// ---------------------------------------------------------------------------
extern "C" void kernel_launch(void* const* d_in, const int* in_sizes, int n_in,
                              void* d_out, int out_size, void* d_ws, size_t ws_size,
                              hipStream_t stream) {
    (void)in_sizes; (void)n_in; (void)out_size; (void)ws_size;

    const float* x    = (const float*)d_in[0];   // (32,2048,512)
    const float* Win  = (const float*)d_in[1];   // (256,512)
    const float* bin  = (const float*)d_in[2];   // (256,)
    const float* V    = (const float*)d_in[3];   // (256,256)
    const float* brec = (const float*)d_in[4];   // (256,)
    float* out = (float*)d_out;                  // (32,2048,256) f32

    float*  proj = (float*)d_ws;                                   // 64 MB
    bf16_t* Vbf  = (bf16_t*)((char*)d_ws + (size_t)BT * HDIM * 4); // 128 KB

    snn_prep_vbf <<<dim3((HDIM * HDIM) / 256), dim3(256), 0, stream>>>(V, Vbf);
    snn_proj_wmma<<<dim3(BT / 16),             dim3(32),  0, stream>>>(x, Win, bin, proj);
    snn_scan     <<<dim3(BATCH / 16),          dim3(256), SPK_BYTES, stream>>>(proj, Vbf, brec, out);
}
